// MMoELayer_36713380446620
// MI455X (gfx1250) — compile-verified, matrix operand order checked by reading, not run on hardware
//
#include <hip/hip_runtime.h>
#include <hip/hip_bf16.h>

// ---------------------------------------------------------------------------
// MMoE fused kernel for gfx1250 (MI455X), wave32 + WMMA bf16.
//   GEMM1: Y[32 x 416] = x_tile[32 x 1024] * Wc[1024 x 416]   (bf16 wmma)
//     cols 0..255  -> experts (e*64+h), bias+ReLU -> h (restaged as A-frags)
//     cols 256..403-> gates (t*4+e), bias -> logits (LDS)
//   GEMM2: ES[32 x 4 x 64] = h * We2 + be2                     (bf16 wmma)
//   softmax over E=4, combine (VALU), write outputs.
// 32 rows/block + ~131KB LDS -> 2 blocks/WGP; 2 independent WMMA chains/wave.
// ---------------------------------------------------------------------------

typedef __attribute__((ext_vector_type(16))) __bf16        v16bf;
typedef __attribute__((ext_vector_type(8)))  float         v8f;
typedef __attribute__((ext_vector_type(8)))  unsigned int  v8u;

#define BB   65536
#define INN  1024
#define EE   4
#define HH   64
#define TT   37
#define NT1  26          // 416/16 N-tiles for GEMM1 (404 real cols, padded)
#define KT1  32          // 1024/32 K-steps
#define ROWS 32          // rows per block
#define MT   2           // 16-row m-tiles per block

// ws layout (bf16 elements)
#define WPACK_ELEMS  (NT1 * KT1 * 32 * 16)      // 425984
#define W2PACK_ELEMS (EE * 4 * 2 * 32 * 16)     // 16384

// LDS layout (bytes)
#define XA_OFF   0
#define XA_SZ    (MT * KT1 * 32 * 32)           // 65536
#define HA_OFF   (XA_OFF + XA_SZ)
#define HA_SZ    (MT * EE * 2 * 32 * 32)        // 16384
#define GL_OFF   (HA_OFF + HA_SZ)
#define GL_SZ    (ROWS * 148 * 4)               // 18944
#define ES_OFF   (GL_OFF + GL_SZ)
#define ES_SZ    (ROWS * EE * HH * 4)           // 32768
#define SMEM_SZ  (ES_OFF + ES_SZ)               // 133632 -> 2 blocks per WGP

__device__ __forceinline__ unsigned short f2bf(float f) {
    unsigned int u = __builtin_bit_cast(unsigned int, f);
    u = (u + 0x7FFFu + ((u >> 16) & 1u)) >> 16;   // round-to-nearest-even
    return (unsigned short)u;
}
__device__ __forceinline__ unsigned int pk2bf(float a, float b) {
    return (unsigned int)f2bf(a) | ((unsigned int)f2bf(b) << 16);
}
// (lane, elem) -> k-local for 16-bit WMMA operand fragment layout
__device__ __forceinline__ int frag_klocal(int lane, int e) {
    return (e & 7) | ((e & 8) << 1) | ((lane >= 16) ? 8 : 0);
}

// ---------------------------------------------------------------------------
// Pack weights into per-lane-contiguous bf16 B-fragment blobs (resident in L2)
// ---------------------------------------------------------------------------
__global__ void mmoe_pack_weights(const float* __restrict__ We1,
                                  const float* __restrict__ We2,
                                  const float* __restrict__ Wg,
                                  unsigned short* __restrict__ wpack,
                                  unsigned short* __restrict__ w2pack) {
    int idx = blockIdx.x * blockDim.x + threadIdx.x;
    if (idx < WPACK_ELEMS) {
        int e    = idx & 15;
        int lane = (idx >> 4) & 31;
        int kt   = (idx >> 9) & 31;
        int nt   = idx >> 14;                       // 0..25
        int k = kt * 32 + frag_klocal(lane, e);     // 0..1023
        int n = nt * 16 + (lane & 15);              // 0..415
        float v = 0.0f;
        if (n < 256) {
            int ei = n >> 6, h = n & 63;
            v = We1[((size_t)ei * INN + k) * HH + h];
        } else if (n < 404) {
            int g = n - 256, t = g >> 2, eg = g & 3;
            v = Wg[((size_t)t * INN + k) * EE + eg];
        }
        wpack[idx] = f2bf(v);
    } else if (idx < WPACK_ELEMS + W2PACK_ELEMS) {
        int j    = idx - WPACK_ELEMS;
        int e    = j & 15;
        int lane = (j >> 4) & 31;
        int kt2  = (j >> 9) & 1;
        int nh   = (j >> 10) & 3;
        int e2   = j >> 12;                         // 0..3
        int k = kt2 * 32 + frag_klocal(lane, e);    // 0..63 (contraction = h)
        int n = nh * 16 + (lane & 15);              // 0..63
        w2pack[j] = f2bf(We2[((size_t)e2 * HH + k) * HH + n]);
    }
}

// GEMM1 epilogue for one 16x16 D tile (role selected by nt)
__device__ __forceinline__ void gemm1_epilogue(int nt, v8f acc, int lane, int mt,
                                               int mbase,
                                               unsigned short* HA, float* GL,
                                               const float* __restrict__ be1,
                                               const float* __restrict__ bg) {
    int c = nt * 16 + (lane & 15);
    if (nt < 16) {
        // experts: bias + ReLU, restage as bf16 A-frags for GEMM2
        int ei = c >> 6, h = c & 63;
        float bias = be1[ei * HH + h];
        int kt2 = h >> 5, kl = h & 31;
        int lp = ((kl >> 3) & 1) << 4;
        int ep = (kl & 7) | ((kl & 16) >> 1);
        unsigned short* hb = HA + ((((mt * EE + ei) * 2 + kt2) << 5) << 4);
#pragma unroll
        for (int r2 = 0; r2 < 8; ++r2) {
            float v = acc[r2] + bias;
            v = v > 0.0f ? v : 0.0f;
            hb[((mbase + r2 + lp) << 4) + ep] = f2bf(v);
        }
    } else {
        // gates: bias, stash logits
        int g = c - 256;
        if (g < 148) {
            int t = g >> 2, eg = g & 3;
            float bias = bg[t * EE + eg];
#pragma unroll
            for (int r2 = 0; r2 < 8; ++r2) {
                GL[(mt * 16 + mbase + r2) * 148 + g] = acc[r2] + bias;
            }
        }
    }
}

// ---------------------------------------------------------------------------
// Main fused kernel: one block = 32 rows, 8 waves (wave32)
// ---------------------------------------------------------------------------
__global__ void __launch_bounds__(256)
mmoe_main(const float* __restrict__ x,
          const float* __restrict__ be1,
          const float* __restrict__ be2,
          const float* __restrict__ bg,
          const unsigned short* __restrict__ wpack,
          const unsigned short* __restrict__ w2pack,
          float* __restrict__ out_task,
          float* __restrict__ out_gate) {
    extern __shared__ char sm[];
    char*           XA  = sm + XA_OFF;
    unsigned short* HA  = (unsigned short*)(sm + HA_OFF);
    float*          GL  = (float*)(sm + GL_OFF);
    float*          ESf = (float*)(sm + ES_OFF);

    const int tid  = threadIdx.x;
    const int lane = tid & 31;
    const int wid  = tid >> 5;
    const int b0   = blockIdx.x * ROWS;

    // ---- Phase 1: stage x tile into LDS as bf16 A-fragments -------------
    // run = 8 consecutive k of one row -> one lane's 16B half-fragment
    for (int i = tid; i < ROWS * (INN / 8); i += 256) {
        int r = i >> 7;                 // row in tile 0..31
        int k = (i & 127) << 3;         // 0,8,...,1016
        const float4* src = (const float4*)(x + (size_t)(b0 + r) * INN + k);
        float4 f0 = src[0];
        float4 f1 = src[1];
        unsigned int d0 = pk2bf(f0.x, f0.y), d1 = pk2bf(f0.z, f0.w);
        unsigned int d2 = pk2bf(f1.x, f1.y), d3 = pk2bf(f1.z, f1.w);
        int mt = r >> 4, mr = r & 15;
        int kt = k >> 5, kl = k & 31;                 // kl in {0,8,16,24}
        int dlane = mr + (((kl >> 3) & 1) << 4);
        int e0    = (kl & 7) | ((kl & 16) >> 1);      // 0 or 8
        uint4* dst = (uint4*)(XA + (((mt * KT1 + kt) << 5) + dlane) * 32 + ((e0 >> 1) << 2));
        *dst = make_uint4(d0, d1, d2, d3);
    }
    __syncthreads();

    // ---- Phase 2: GEMM1 (x * Wc), 2 N-tiles per pass sharing A ----------
    const int mt    = wid & 1;
    const int nrole = wid >> 1;               // 0..3
    // tile ranges: experts 0..7 / 8..15, gates 16..21 / 22..25 (all even)
    const int ntLo  = (nrole == 0) ? 0 : (nrole == 1) ? 8 : (nrole == 2) ? 16 : 22;
    const int ntHi  = (nrole == 0) ? 8 : (nrole == 1) ? 16 : (nrole == 2) ? 22 : 26;
    const int mbase = (lane >> 4) << 3;       // 0 or 8

    for (int nt = ntLo; nt < ntHi; nt += 2) {
        v8f acc0 = {};
        v8f acc1 = {};
        for (int kt = 0; kt < KT1; ++kt) {
            const v8u* ap  = (const v8u*)(XA + (((mt * KT1 + kt) << 5) + lane) * 32);
            const v8u* bp0 = (const v8u*)((const char*)wpack +
                                          (((((nt + 0) * KT1 + kt) << 5) + lane) << 5));
            const v8u* bp1 = (const v8u*)((const char*)wpack +
                                          (((((nt + 1) * KT1 + kt) << 5) + lane) << 5));
            v16bf a  = __builtin_bit_cast(v16bf, *ap);
            v16bf b0 = __builtin_bit_cast(v16bf, *bp0);
            v16bf b1 = __builtin_bit_cast(v16bf, *bp1);
            acc0 = __builtin_amdgcn_wmma_f32_16x16x32_bf16(
                       false, a, false, b0, (short)0, acc0, false, false);
            acc1 = __builtin_amdgcn_wmma_f32_16x16x32_bf16(
                       false, a, false, b1, (short)0, acc1, false, false);
        }
        gemm1_epilogue(nt + 0, acc0, lane, mt, mbase, HA, GL, be1, bg);
        gemm1_epilogue(nt + 1, acc1, lane, mt, mbase, HA, GL, be1, bg);
    }
    __syncthreads();

    // ---- Phase 3a: GEMM2 (h * We2 + be2) -> expert_stack in LDS ---------
    // wave -> (mt2, e); A-frag shared across 4 independent H-tiles
    {
        const int mt2 = wid & 1;
        const int e   = wid >> 1;             // 0..3
        v8f acc[4] = {{}, {}, {}, {}};
#pragma unroll
        for (int kt2 = 0; kt2 < 2; ++kt2) {
            const v8u* ap = (const v8u*)((const char*)HA +
                            (((((mt2 * EE + e) * 2 + kt2) << 5) + lane) << 5));
            v16bf a = __builtin_bit_cast(v16bf, *ap);
#pragma unroll
            for (int nh = 0; nh < 4; ++nh) {
                const v8u* bp = (const v8u*)((const char*)w2pack +
                                (((((e * 4 + nh) * 2 + kt2) << 5) + lane) << 5));
                v16bf b = __builtin_bit_cast(v16bf, *bp);
                acc[nh] = __builtin_amdgcn_wmma_f32_16x16x32_bf16(
                              false, a, false, b, (short)0, acc[nh], false, false);
            }
        }
#pragma unroll
        for (int nh = 0; nh < 4; ++nh) {
            int hc = nh * 16 + (lane & 15);
            float bias = be2[e * HH + hc];
#pragma unroll
            for (int r2 = 0; r2 < 8; ++r2) {
                ESf[((mt2 * 16 + mbase + r2) * EE + e) * HH + hc] = acc[nh][r2] + bias;
            }
        }
    }

    // ---- Phase 3b: softmax over E=4 per (row, task), write gate output --
    for (int i = tid; i < ROWS * TT; i += 256) {
        int row = i / TT;
        int t   = i - row * TT;
        float* gp = GL + row * 148 + t * 4;
        float a0 = gp[0], a1 = gp[1], a2 = gp[2], a3 = gp[3];
        float mx = fmaxf(fmaxf(a0, a1), fmaxf(a2, a3));
        float w0 = __expf(a0 - mx), w1 = __expf(a1 - mx);
        float w2 = __expf(a2 - mx), w3 = __expf(a3 - mx);
        float inv = 1.0f / (w0 + w1 + w2 + w3);
        w0 *= inv; w1 *= inv; w2 *= inv; w3 *= inv;
        gp[0] = w0; gp[1] = w1; gp[2] = w2; gp[3] = w3;
        float* og = out_gate + ((size_t)(b0 + row) * TT + t) * EE;
        og[0] = w0; og[1] = w1; og[2] = w2; og[3] = w3;
    }
    __syncthreads();

    // ---- Phase 4: combine task_outputs[b][t][:] = sum_e g * ES[b][e][:] -
    for (int i = tid; i < ROWS * TT; i += 256) {
        int row = i / TT;
        int t   = i - row * TT;
        const float* gp = GL + row * 148 + t * 4;
        float g0 = gp[0], g1 = gp[1], g2 = gp[2], g3 = gp[3];
        const float4* e0p = (const float4*)(ESf + (row * EE + 0) * HH);
        const float4* e1p = (const float4*)(ESf + (row * EE + 1) * HH);
        const float4* e2p = (const float4*)(ESf + (row * EE + 2) * HH);
        const float4* e3p = (const float4*)(ESf + (row * EE + 3) * HH);
        float4* op = (float4*)(out_task + ((size_t)(b0 + row) * TT + t) * HH);
#pragma unroll 4
        for (int hq = 0; hq < HH / 4; ++hq) {
            float4 v0 = e0p[hq], v1 = e1p[hq], v2 = e2p[hq], v3 = e3p[hq];
            float4 o;
            o.x = g0 * v0.x + g1 * v1.x + g2 * v2.x + g3 * v3.x;
            o.y = g0 * v0.y + g1 * v1.y + g2 * v2.y + g3 * v3.y;
            o.z = g0 * v0.z + g1 * v1.z + g2 * v2.z + g3 * v3.z;
            o.w = g0 * v0.w + g1 * v1.w + g2 * v2.w + g3 * v3.w;
            op[hq] = o;
        }
    }
}

// ---------------------------------------------------------------------------
extern "C" void kernel_launch(void* const* d_in, const int* in_sizes, int n_in,
                              void* d_out, int out_size, void* d_ws, size_t ws_size,
                              hipStream_t stream) {
    const float* x   = (const float*)d_in[0];
    const float* We1 = (const float*)d_in[1];
    const float* be1 = (const float*)d_in[2];
    const float* We2 = (const float*)d_in[3];
    const float* be2 = (const float*)d_in[4];
    const float* Wg  = (const float*)d_in[5];
    const float* bg  = (const float*)d_in[6];

    unsigned short* wpack  = (unsigned short*)d_ws;
    unsigned short* w2pack = wpack + WPACK_ELEMS;

    float* out_task = (float*)d_out;
    float* out_gate = out_task + (size_t)BB * TT * HH;

    // pack weights (bf16 fragment blobs), tiny: ~0.9 MB -> lives in L2
    {
        int total  = WPACK_ELEMS + W2PACK_ELEMS;      // 442368
        int blocks = (total + 255) / 256;             // 1728
        mmoe_pack_weights<<<blocks, 256, 0, stream>>>(We1, We2, Wg, wpack, w2pack);
    }

    // main fused kernel: 32 rows per block
    (void)hipFuncSetAttribute((const void*)mmoe_main,
                              hipFuncAttributeMaxDynamicSharedMemorySize, SMEM_SZ);
    mmoe_main<<<BB / ROWS, 256, SMEM_SZ, stream>>>(x, be1, be2, bg,
                                                   wpack, w2pack, out_task, out_gate);
}
